// Head_49950469652969
// MI455X (gfx1250) — compile-verified, matrix operand order checked by reading, not run on hardware
//
#include <hip/hip_runtime.h>

typedef __bf16 bf16;
typedef __attribute__((ext_vector_type(16))) __bf16 v16bf;
typedef __attribute__((ext_vector_type(8)))  float  v8f;

union AFrag { v16bf v; unsigned int u[8]; };

static __device__ __forceinline__ unsigned int pack2(float a, float b) {
  union { unsigned int u; __bf16 h[2]; } t;
  t.h[0] = (__bf16)a; t.h[1] = (__bf16)b;
  return t.u;
}

// Use CDNA5 async global->LDS copies when the toolchain exposes them.
#if defined(__has_builtin)
#if __has_builtin(__builtin_amdgcn_global_load_async_to_lds_b64) && \
    __has_builtin(__builtin_amdgcn_s_wait_asynccnt)
#define USE_ASYNC_LDS 1
#endif
#endif

// pointee type of the async b64 builtin (from the compiler diagnostic):
// int __attribute__((vector_size(8)))
typedef int async_b64_t __attribute__((vector_size(8)));

// ---------------------------------------------------------------------------
// Kernel 1: fused K/Q/V projection.  out = x @ W^T  (bf16 outputs)
// grid = (BT/16, 3), block = 256 (8 waves).  Block computes a 16x128 tile of
// one projection; wave w owns columns [16w, 16w+16).  The fp32 x tile is
// staged in LDS via async global->LDS copies (ASYNCcnt) when available.
// Projection 2 (V) is written transposed: Vt[b][h][t] so that attention can
// load packed bf16 pairs along t for its B-fragments.
// ---------------------------------------------------------------------------
__global__ __launch_bounds__(256) void proj_kernel(
    const float* __restrict__ x,  const float* __restrict__ Wk,
    const float* __restrict__ Wq, const float* __restrict__ Wv,
    bf16* __restrict__ Kb, bf16* __restrict__ Qb, bf16* __restrict__ Vt)
{
  constexpr int C = 1024, H = 128, T = 2048;
  __shared__ float xtf[16 * 32];

  const float* W;
  if (blockIdx.y == 0)      W = Wk;
  else if (blockIdx.y == 1) W = Wq;
  else                      W = Wv;

  const int row0 = blockIdx.x * 16;
  const int lane = threadIdx.x & 31;
  const int wv   = threadIdx.x >> 5;   // 0..7 -> column tile
  const int half = lane >> 4;
  const int l16  = lane & 15;

  v8f acc = {};
  for (int k0 = 0; k0 < C; k0 += 32) {
    // stage x[row0:+16, k0:+32] (fp32) into LDS: 8 bytes per thread
    const int r = threadIdx.x >> 4;        // 0..15
    const int c = (threadIdx.x & 15) * 2;  // 0..30
    const float* gsrc = x + (size_t)(row0 + r) * C + (k0 + c);
    float* ldst = xtf + r * 32 + c;
#ifdef USE_ASYNC_LDS
    __builtin_amdgcn_global_load_async_to_lds_b64(
        (__attribute__((address_space(1))) async_b64_t*)gsrc,
        (__attribute__((address_space(3))) async_b64_t*)ldst, 0, 0);
    __builtin_amdgcn_s_wait_asynccnt(0);
#else
    float2 xv = *(const float2*)gsrc;
    ldst[0] = xv.x;
    ldst[1] = xv.y;
#endif
    __syncthreads();

    AFrag a, b;
#pragma unroll
    for (int j = 0; j < 8; ++j) {
      const int kk = (j >> 2) * 16 + half * 8 + (j & 3) * 2;
      a.u[j] = pack2(xtf[l16 * 32 + kk], xtf[l16 * 32 + kk + 1]);
      const int n = wv * 16 + l16;
      float2 wl = *(const float2*)(W + (size_t)n * C + (k0 + kk));
      b.u[j] = pack2(wl.x, wl.y);
    }
    acc = __builtin_amdgcn_wmma_f32_16x16x32_bf16(false, a.v, false, b.v,
                                                  (short)0, acc, false, false);
    __syncthreads();
  }

  if (blockIdx.y < 2) {
    bf16* O = (blockIdx.y == 0) ? Kb : Qb;
#pragma unroll
    for (int i = 0; i < 8; ++i) {
      const int row = row0 + i + 8 * half;
      O[(size_t)row * H + wv * 16 + l16] = (bf16)acc[i];
    }
  } else {
    const int bidx = row0 / T;
    const int t0   = row0 % T;
#pragma unroll
    for (int i = 0; i < 8; ++i)
      Vt[((size_t)bidx * H + wv * 16 + l16) * T + (t0 + i + 8 * half)] =
          (bf16)acc[i];
  }
}

// ---------------------------------------------------------------------------
// Kernel 2: causal flash attention.  One wave per 16-query tile; block = 8
// waves; grid = B*T/16/8 = 128 blocks.  Q held in registers (4 A-frags),
// key tiles of 32 processed with online softmax; P transposed C->A layout
// through per-wave LDS scratch (s_wait_dscnt for same-wave ordering since
// waves have divergent causal trip counts).  Next K/V tiles prefetched.
// ---------------------------------------------------------------------------
__global__ __launch_bounds__(256) void attn_kernel(
    const bf16* __restrict__ Kb, const bf16* __restrict__ Qb,
    const bf16* __restrict__ Vt, float* __restrict__ out)
{
  constexpr int   T = 2048, H = 128;
  constexpr float scale = 0.03125f;  // C**-0.5, C=1024 (faithful to reference)
  __shared__ bf16 psh[8][16 * 32];

  const int lane = threadIdx.x & 31;
  const int wv   = threadIdx.x >> 5;
  const int half = lane >> 4;
  const int l16  = lane & 15;
  const int tile = blockIdx.x * 8 + wv;   // 0..1023
  const int bidx = tile >> 7;
  const int qb   = (tile & 127) * 16;

  const bf16* Kp = Kb + (size_t)bidx * T * H;
  const bf16* Qp = Qb + (size_t)bidx * T * H;
  const bf16* Vp = Vt + (size_t)bidx * H * T;

  // Q fragments: 16 rows x 128 (4 k-slices of 32)
  AFrag qf[4];
#pragma unroll
  for (int hk = 0; hk < 4; ++hk)
#pragma unroll
    for (int j = 0; j < 8; ++j) {
      const int kk = hk * 32 + (j >> 2) * 16 + half * 8 + (j & 3) * 2;
      qf[hk].u[j] = *(const unsigned int*)(Qp + (size_t)(qb + l16) * H + kk);
    }

  v8f   o[8] = {};
  float m[8], lsum[8];
#pragma unroll
  for (int i = 0; i < 8; ++i) { m[i] = -3.0e38f; lsum[i] = 0.0f; }

  const int nkt = (qb + 15) / 32 + 1;  // causal: key tiles of 32
  for (int kt = 0; kt < nkt; ++kt) {
    const int kb = kt * 32;

    // prefetch next key/value tiles while this one computes
    if (kt + 1 < nkt) {
      __builtin_prefetch(Kp + (size_t)(kb + 32 + lane) * H, 0, 1);
      __builtin_prefetch(Vp + (size_t)(lane * 4) * T + (kb + 32), 0, 1);
    }

    // S = Q K^T : two 16x16 tiles (keys kb..kb+15 and kb+16..kb+31)
    v8f s0 = {}, s1 = {};
#pragma unroll
    for (int hk = 0; hk < 4; ++hk) {
      AFrag b0, b1;
#pragma unroll
      for (int j = 0; j < 8; ++j) {
        const int kk = hk * 32 + (j >> 2) * 16 + half * 8 + (j & 3) * 2;
        b0.u[j] = *(const unsigned int*)(Kp + (size_t)(kb + l16) * H + kk);
        b1.u[j] = *(const unsigned int*)(Kp + (size_t)(kb + 16 + l16) * H + kk);
      }
      s0 = __builtin_amdgcn_wmma_f32_16x16x32_bf16(false, qf[hk].v, false, b0.v,
                                                   (short)0, s0, false, false);
      s1 = __builtin_amdgcn_wmma_f32_16x16x32_bf16(false, qf[hk].v, false, b1.v,
                                                   (short)0, s1, false, false);
    }

    // online softmax update (row stats live in the 16 lanes of a half-wave)
#pragma unroll
    for (int i = 0; i < 8; ++i) {
      const int r  = i + 8 * half;
      const int qg = qb + r;
      float x0 = s0[i] * scale;
      float x1 = s1[i] * scale;
      if (kb + l16 > qg)      x0 = -3.0e38f;
      if (kb + 16 + l16 > qg) x1 = -3.0e38f;
      float mx = fmaxf(x0, x1);
#pragma unroll
      for (int d = 1; d < 16; d <<= 1) mx = fmaxf(mx, __shfl_xor(mx, d, 32));
      const float mn    = fmaxf(m[i], mx);
      const float p0    = __expf(x0 - mn);
      const float p1    = __expf(x1 - mn);
      const float alpha = __expf(m[i] - mn);
      float rs = p0 + p1;
#pragma unroll
      for (int d = 1; d < 16; d <<= 1) rs += __shfl_xor(rs, d, 32);
      lsum[i] = lsum[i] * alpha + rs;
      m[i]    = mn;
#pragma unroll
      for (int ht = 0; ht < 8; ++ht) o[ht][i] = o[ht][i] * alpha;
      // stash P (C-layout) into per-wave LDS scratch as bf16
      psh[wv][r * 32 + l16]      = (bf16)p0;
      psh[wv][r * 32 + 16 + l16] = (bf16)p1;
    }
    asm volatile("s_wait_dscnt 0" ::: "memory");  // same-wave LDS RAW fence

    // reload P in A-layout (16 queries x 32 keys)
    AFrag pa;
#pragma unroll
    for (int j = 0; j < 8; ++j) {
      const int kk = (j >> 2) * 16 + half * 8 + (j & 3) * 2;
      pa.u[j] = *(const unsigned int*)(&psh[wv][l16 * 32 + kk]);
    }

    // O += P V : 8 column tiles of H, V read from transposed layout
#pragma unroll
    for (int ht = 0; ht < 8; ++ht) {
      AFrag vb;
#pragma unroll
      for (int j = 0; j < 8; ++j) {
        const int kk = (j >> 2) * 16 + half * 8 + (j & 3) * 2;
        vb.u[j] = *(const unsigned int*)(
            Vp + (size_t)(ht * 16 + l16) * T + (kb + kk));
      }
      o[ht] = __builtin_amdgcn_wmma_f32_16x16x32_bf16(false, pa.v, false, vb.v,
                                                      (short)0, o[ht], false,
                                                      false);
    }
  }

  // normalize and store fp32 output [B,T,H]
#pragma unroll
  for (int ht = 0; ht < 8; ++ht)
#pragma unroll
    for (int i = 0; i < 8; ++i) {
      const int r = i + 8 * half;
      out[((size_t)bidx * T + qb + r) * H + ht * 16 + l16] =
          o[ht][i] / lsum[i];
    }
}

// ---------------------------------------------------------------------------
extern "C" void kernel_launch(void* const* d_in, const int* in_sizes, int n_in,
                              void* d_out, int out_size, void* d_ws,
                              size_t ws_size, hipStream_t stream) {
  (void)in_sizes; (void)n_in; (void)out_size; (void)ws_size;
  const float* x  = (const float*)d_in[0];
  const float* Wk = (const float*)d_in[1];
  const float* Wq = (const float*)d_in[2];
  const float* Wv = (const float*)d_in[3];
  float* out = (float*)d_out;

  const size_t BT        = (size_t)8 * 2048;           // 16384 rows
  const size_t projBytes = BT * 128 * sizeof(short);   // 4 MB per tensor
  bf16* Kb = (bf16*)d_ws;
  bf16* Qb = (bf16*)((char*)d_ws + projBytes);
  bf16* Vt = (bf16*)((char*)d_ws + 2 * projBytes);

  dim3 g1((unsigned)(BT / 16), 3), b1(256);
  proj_kernel<<<g1, b1, 0, stream>>>(x, Wk, Wq, Wv, Kb, Qb, Vt);

  dim3 g2(128), b2(256);
  attn_kernel<<<g2, b2, 0, stream>>>(Kb, Qb, Vt, out);
}
